// DiTBlock_44083544326972
// MI455X (gfx1250) — matrix operand order repaired and checked by reading, not compiled, not run on hardware
//
#include <hip/hip_runtime.h>
#include <hip/hip_bf16.h>

// ---------------- problem constants ----------------
#define Bc   8
#define Sc   2048
#define Dc   1024
#define Hc   16
#define Kc   256
#define DHc  64
#define MLPc 4096
#define ZDc  1024

typedef __bf16 bf16;
typedef __bf16 v16bf __attribute__((ext_vector_type(16)));
typedef float  v8f   __attribute__((ext_vector_type(8)));

// ---------------- WMMA helpers ----------------
__device__ __forceinline__ v8f wmma_bf16(v16bf a, v16bf b, v8f c) {
  // D = A(16x32 bf16) * B(32x16 bf16) + C(16x16 f32)
  return __builtin_amdgcn_wmma_f32_16x16x32_bf16(false, a, false, b, (short)0, c,
                                                 false, false);
}

__device__ __forceinline__ v8f zero8() {
  v8f z;
#pragma unroll
  for (int i = 0; i < 8; ++i) z[i] = 0.0f;
  return z;
}

// Per-lane async copy of 16 bytes global -> LDS (ASYNCcnt-tracked TDM-lite path).
// ldsOff: low 32 bits of a generic __shared__ pointer == LDS byte offset.
__device__ __forceinline__ void async_g2l_b128(unsigned ldsOff, const void* sbase,
                                               unsigned gByteOff) {
  asm volatile("global_load_async_to_lds_b128 %0, %1, %2"
               :: "v"(ldsOff), "v"(gByteOff), "s"(sbase)
               : "memory");
}
__device__ __forceinline__ void wait_async0() {
  asm volatile("s_wait_asynccnt 0x0" ::: "memory");
}

// A fragment: 16x32 bf16 tile, LDS row-major [m][k] with element stride `stride`.
// Lane l(0..15): row m=l, K={kh..kh+7, 16+kh..16+kh+7}, kh = (l>>4)*8.
__device__ __forceinline__ v16bf frag_a(const bf16* base, int stride, int m, int k0) {
  const int lane = threadIdx.x & 31;
  const bf16* p = base + (m + (lane & 15)) * stride + k0 + ((lane >> 4) << 3);
  v16bf a;
#pragma unroll
  for (int j = 0; j < 8; ++j) { a[j] = p[j]; a[8 + j] = p[16 + j]; }
  return a;
}

// B fragment: 32x16 bf16 tile, LDS stored TRANSPOSED [n][k] with stride `stride`.
// Lane l(0..15): col n=l, K={kh..kh+7, 16+kh..16+kh+7}, kh = (l>>4)*8.
__device__ __forceinline__ v16bf frag_b(const bf16* base, int stride, int n, int k0) {
  const int lane = threadIdx.x & 31;
  const bf16* p = base + (n + (lane & 15)) * stride + k0 + ((lane >> 4) << 3);
  v16bf b;
#pragma unroll
  for (int j = 0; j < 8; ++j) { b[j] = p[j]; b[8 + j] = p[16 + j]; }
  return b;
}

__device__ __forceinline__ float gelu_exact(float x) {
  return 0.5f * x * (1.0f + erff(x * 0.70710678118654752f));
}

// ---------------- f32 -> bf16 convert ----------------
__global__ __launch_bounds__(256) void cvt_bf16_kernel(const float* __restrict__ in,
                                                       bf16* __restrict__ out, int n4) {
  int i = (blockIdx.x * 256 + threadIdx.x);
  if (i < n4) {
    float4 v = *(const float4*)(in + (size_t)i * 4);
    bf16* o = out + (size_t)i * 4;
    o[0] = (bf16)v.x; o[1] = (bf16)v.y; o[2] = (bf16)v.z; o[3] = (bf16)v.w;
  }
}

// ---------------- tiny conditioning GEMMs (B=8 rows) ----------------
__global__ __launch_bounds__(256) void small_dense_kernel(
    const float* __restrict__ inp, const float* __restrict__ W,
    const float* __restrict__ bias, float* __restrict__ out, int Kd, int N, int act) {
  int b = blockIdx.y;
  int n = blockIdx.x * 256 + threadIdx.x;
  const float* xr = inp + (size_t)b * Kd;
  float acc = bias[n];
  for (int k = 0; k < Kd; ++k) acc = fmaf(xr[k], W[(size_t)k * N + n], acc);
  if (act) acc = acc / (1.0f + __expf(-acc));  // silu
  out[(size_t)b * N + n] = acc;
}

// ---------------- adaLN: normalize + modulate, write bf16 ----------------
__global__ __launch_bounds__(256) void adaln_kernel(const float* __restrict__ x,
                                                    const float* __restrict__ scale,
                                                    const float* __restrict__ offset,
                                                    bf16* __restrict__ out) {
  __shared__ float s1[256];
  __shared__ float s2[256];
  const int row = blockIdx.x;
  const int b = row / Sc;
  const int tid = threadIdx.x;
  const float* xr = x + (size_t)row * Dc;
  float4 v = *(const float4*)(xr + tid * 4);
  s1[tid] = v.x + v.y + v.z + v.w;
  s2[tid] = v.x * v.x + v.y * v.y + v.z * v.z + v.w * v.w;
  __syncthreads();
#pragma unroll
  for (int st = 128; st > 0; st >>= 1) {
    if (tid < st) { s1[tid] += s1[tid + st]; s2[tid] += s2[tid + st]; }
    __syncthreads();
  }
  const float mean = s1[0] * (1.0f / Dc);
  const float var  = s2[0] * (1.0f / Dc) - mean * mean;
  const float rstd = rsqrtf(var + 1e-6f);
  const int d = tid * 4;
  const float* sc = scale  + (size_t)b * Dc + d;
  const float* of = offset + (size_t)b * Dc + d;
  bf16* o = out + (size_t)row * Dc + d;
  o[0] = (bf16)((v.x - mean) * rstd * sc[0] + of[0]);
  o[1] = (bf16)((v.y - mean) * rstd * sc[1] + of[1]);
  o[2] = (bf16)((v.z - mean) * rstd * sc[2] + of[2]);
  o[3] = (bf16)((v.w - mean) * rstd * sc[3] + of[3]);
}

// ---------------- main bf16 WMMA GEMM: C[M,N] = A[M,K] @ B[K,N] ----------------
// EPI 0: bf16 out, +bias      EPI 1: bf16 out, +bias, exact GELU
// EPI 2: f32 out,  +bias, +residual
template <int EPI>
__global__ __launch_bounds__(256) void gemm_bf16_kernel(
    const bf16* __restrict__ A, const bf16* __restrict__ Bg,
    const float* __restrict__ bias, const float* __restrict__ resid,
    void* __restrict__ Cout, int M, int N, int Kd) {
  __shared__ bf16 As[2][128 * 40];   // [m][k], stride 40
  __shared__ bf16 Bs[2][128 * 40];   // transposed [n][k], stride 40
  const int tid = threadIdx.x;
  const int mTile = blockIdx.y * 128;
  const int nTile = blockIdx.x * 128;
  const int w = tid >> 5;
  const int mBase = (w & 3) * 32;
  const int nBase = (w >> 2) * 64;

  v8f acc[2][4];
#pragma unroll
  for (int mi = 0; mi < 2; ++mi)
#pragma unroll
    for (int ni = 0; ni < 4; ++ni) acc[mi][ni] = zero8();

  // A tile 128x32 via async global->LDS DMA (row-major, no transpose needed)
  auto loadA = [&](int k0, int buf) {
#pragma unroll
    for (int i = 0; i < 2; ++i) {
      int u = tid + i * 256;
      int row = u >> 2, c = u & 3;
      unsigned gOff = (unsigned)(((size_t)(mTile + row) * Kd + k0 + c * 8) * sizeof(bf16));
      unsigned lOff = (unsigned)(uintptr_t)&As[buf][row * 40 + c * 8];
      async_g2l_b128(lOff, A, gOff);
    }
  };
  // B tile 32x128, stored transposed [n][k] (scatter through VGPRs)
  auto loadB = [&](int k0, int buf) {
#pragma unroll
    for (int i = 0; i < 2; ++i) {
      int u = tid + i * 256;
      int kk = u >> 4, c = u & 15;
      union { uint4 v; bf16 h8[8]; } t;
      t.v = *(const uint4*)(Bg + (size_t)(k0 + kk) * N + nTile + c * 8);
#pragma unroll
      for (int e = 0; e < 8; ++e) Bs[buf][(c * 8 + e) * 40 + kk] = t.h8[e];
    }
  };

  const int nk = Kd >> 5;
  loadA(0, 0);
  loadB(0, 0);
  for (int kt = 0; kt < nk; ++kt) {
    const int cur = kt & 1;
    wait_async0();       // my async A-tile writes for `cur` have landed
    __syncthreads();     // everyone's A(async) + B(ds) writes visible
    if (kt + 1 < nk) {
      loadA((kt + 1) * 32, cur ^ 1);
      loadB((kt + 1) * 32, cur ^ 1);
    }
    if (kt + 2 < nk)  // warm L2 for next-next A tile
      __builtin_prefetch(A + (size_t)(mTile + (tid >> 1)) * Kd + (size_t)(kt + 2) * 32, 0, 1);
    v16bf af[2], bfr[4];
#pragma unroll
    for (int mi = 0; mi < 2; ++mi) af[mi] = frag_a(As[cur], 40, mBase + 16 * mi, 0);
#pragma unroll
    for (int ni = 0; ni < 4; ++ni) bfr[ni] = frag_b(Bs[cur], 40, nBase + 16 * ni, 0);
#pragma unroll
    for (int mi = 0; mi < 2; ++mi)
#pragma unroll
      for (int ni = 0; ni < 4; ++ni) acc[mi][ni] = wmma_bf16(af[mi], bfr[ni], acc[mi][ni]);
  }

  const int lane = tid & 31;
#pragma unroll
  for (int mi = 0; mi < 2; ++mi) {
#pragma unroll
    for (int ni = 0; ni < 4; ++ni) {
      const int col  = nTile + nBase + ni * 16 + (lane & 15);
      const int row0 = mTile + mBase + mi * 16 + ((lane >> 4) << 3);
      const float bcol = bias[col];
#pragma unroll
      for (int r = 0; r < 8; ++r) {
        const size_t idx = (size_t)(row0 + r) * N + col;
        float v = acc[mi][ni][r] + bcol;
        if constexpr (EPI == 1) v = gelu_exact(v);
        if constexpr (EPI == 2) {
          ((float*)Cout)[idx] = v + resid[idx];
        } else {
          ((bf16*)Cout)[idx] = (bf16)v;
        }
      }
    }
  }
}

// ---------------- Linformer seq projection ----------------
// per z=(b*H+h): Out[z, m(0..255), n(0..63)] = sum_s Ew[s,m] * X[b,s,h*64+n] + biasM[m]
__global__ __launch_bounds__(256) void seqproj_kernel(
    const bf16* __restrict__ Ewb, const bf16* __restrict__ X,
    const float* __restrict__ biasM, bf16* __restrict__ Out) {
  __shared__ bf16 As[2][128 * 40];  // [m][k]
  __shared__ bf16 Bs[2][64 * 40];   // transposed [n][k]
  const int tid = threadIdx.x;
  const int z = blockIdx.y;
  const int b = z >> 4, h = z & 15;
  const int mTile = blockIdx.x * 128;
  const size_t xBase = (size_t)b * Sc * Dc + h * DHc;
  const int w = tid >> 5;
  const int mBase = (w >> 1) * 32;
  const int nBase = (w & 1) * 32;

  v8f acc[2][2];
#pragma unroll
  for (int mi = 0; mi < 2; ++mi)
#pragma unroll
    for (int ni = 0; ni < 2; ++ni) acc[mi][ni] = zero8();

  auto loadT = [&](int k0, int buf) {
    // A = Ew^T : As[m][kk] = Ew[(k0+kk)*Kc + mTile + m]
#pragma unroll
    for (int i = 0; i < 2; ++i) {
      int u = tid + i * 256;
      int kk = u >> 4, c = u & 15;
      union { uint4 v; bf16 h8[8]; } t;
      t.v = *(const uint4*)(Ewb + (size_t)(k0 + kk) * Kc + mTile + c * 8);
#pragma unroll
      for (int e = 0; e < 8; ++e) As[buf][(c * 8 + e) * 40 + kk] = t.h8[e];
    }
    // B: Bs[n][kk] = X[xBase + (k0+kk)*Dc + n]
    {
      int kk = tid >> 3, c = tid & 7;
      union { uint4 v; bf16 h8[8]; } t;
      t.v = *(const uint4*)(X + xBase + (size_t)(k0 + kk) * Dc + c * 8);
#pragma unroll
      for (int e = 0; e < 8; ++e) Bs[buf][(c * 8 + e) * 40 + kk] = t.h8[e];
    }
  };

  const int nk = Sc / 32;
  loadT(0, 0);
  __syncthreads();
  for (int kt = 0; kt < nk; ++kt) {
    const int cur = kt & 1;
    if (kt + 1 < nk) loadT((kt + 1) * 32, cur ^ 1);
    v16bf af[2], bfr[2];
#pragma unroll
    for (int mi = 0; mi < 2; ++mi) af[mi] = frag_a(As[cur], 40, mBase + 16 * mi, 0);
#pragma unroll
    for (int ni = 0; ni < 2; ++ni) bfr[ni] = frag_b(Bs[cur], 40, nBase + 16 * ni, 0);
#pragma unroll
    for (int mi = 0; mi < 2; ++mi)
#pragma unroll
      for (int ni = 0; ni < 2; ++ni) acc[mi][ni] = wmma_bf16(af[mi], bfr[ni], acc[mi][ni]);
    __syncthreads();
  }

  const int lane = tid & 31;
  const size_t oBase = (size_t)z * Kc * DHc;
#pragma unroll
  for (int mi = 0; mi < 2; ++mi) {
#pragma unroll
    for (int ni = 0; ni < 2; ++ni) {
      const int col  = nBase + ni * 16 + (lane & 15);
      const int row0 = mTile + mBase + mi * 16 + ((lane >> 4) << 3);
#pragma unroll
      for (int r = 0; r < 8; ++r) {
        const int row = row0 + r;
        Out[oBase + (size_t)row * DHc + col] = (bf16)(acc[mi][ni][r] + biasM[row]);
      }
    }
  }
}

// ---------------- fused Linformer attention ----------------
// per (b,h,s-tile of 64): O = softmax(Q Kp^T / 8) Vp   (Kp,Vp: [Kc, DHc])
__global__ __launch_bounds__(256) void attn_kernel(const bf16* __restrict__ Q,
                                                   const bf16* __restrict__ KP,
                                                   const bf16* __restrict__ VP,
                                                   bf16* __restrict__ O) {
  __shared__ bf16 Qs[64 * 72];     // [s][dh]
  __shared__ bf16 Ks[256 * 72];    // [kIdx][dh]  (B operand of GEMM1, pairs along dh)
  __shared__ bf16 Vt[64 * 264];    // transposed [dh][kIdx] (B operand of GEMM2)
  __shared__ bf16 Ps[64 * 264];    // softmax probs [s][kIdx]
  __shared__ float redm[64][4];
  __shared__ float reds[64][4];
  const int tid = threadIdx.x;
  const int z = blockIdx.y;
  const int b = z >> 4, h = z & 15;
  const int s0 = blockIdx.x * 64;
  const size_t qBase = (size_t)(b * Sc + s0) * Dc + h * DHc;
  const size_t kvBase = (size_t)z * Kc * DHc;

  // ---- stage tiles into LDS: Q,K via async DMA, V via transpose-scatter ----
#pragma unroll
  for (int i = 0; i < 2; ++i) {
    int u = tid + i * 256;
    int row = u >> 3, c = u & 7;
    unsigned gOff = (unsigned)((qBase + (size_t)row * Dc + c * 8) * sizeof(bf16));
    async_g2l_b128((unsigned)(uintptr_t)(Qs + row * 72 + c * 8), Q, gOff);
  }
#pragma unroll
  for (int i = 0; i < 8; ++i) {
    int u = tid + i * 256;
    int row = u >> 3, c = u & 7;
    unsigned gOff = (unsigned)((kvBase + (size_t)row * DHc + c * 8) * sizeof(bf16));
    async_g2l_b128((unsigned)(uintptr_t)(Ks + row * 72 + c * 8), KP, gOff);
  }
#pragma unroll
  for (int i = 0; i < 8; ++i) {
    int u = tid + i * 256;
    int row = u >> 3, c = u & 7;     // row = kIdx
    union { uint4 v; bf16 h8[8]; } t;
    t.v = *(const uint4*)(VP + kvBase + (size_t)row * DHc + c * 8);
#pragma unroll
    for (int e = 0; e < 8; ++e) Vt[(c * 8 + e) * 264 + row] = t.h8[e];
  }
  wait_async0();
  __syncthreads();

  const int w = tid >> 5, lane = tid & 31;

  // ---- GEMM1: scores 64x256, waves 2(M) x 4(N) ----
  const int m1 = (w & 1) * 32;
  const int n1 = (w >> 1) * 64;
  v8f acc[2][4];
#pragma unroll
  for (int mi = 0; mi < 2; ++mi)
#pragma unroll
    for (int ni = 0; ni < 4; ++ni) acc[mi][ni] = zero8();
#pragma unroll
  for (int ks = 0; ks < 64; ks += 32) {
    v16bf af[2], bfr[4];
#pragma unroll
    for (int mi = 0; mi < 2; ++mi) af[mi] = frag_a(Qs, 72, m1 + 16 * mi, ks);
#pragma unroll
    for (int ni = 0; ni < 4; ++ni) bfr[ni] = frag_b(Ks, 72, n1 + 16 * ni, ks);
#pragma unroll
    for (int mi = 0; mi < 2; ++mi)
#pragma unroll
      for (int ni = 0; ni < 4; ++ni) acc[mi][ni] = wmma_bf16(af[mi], bfr[ni], acc[mi][ni]);
  }

  // ---- softmax over k (cols), rows distributed across 4 N-waves ----
  const float lsc = 0.125f * 1.4426950408889634f;  // (1/sqrt(dh)) * log2(e)
  float rm[2][8];
#pragma unroll
  for (int mi = 0; mi < 2; ++mi)
#pragma unroll
    for (int r = 0; r < 8; ++r)
      rm[mi][r] = fmaxf(fmaxf(acc[mi][0][r], acc[mi][1][r]),
                        fmaxf(acc[mi][2][r], acc[mi][3][r]));
#pragma unroll
  for (int mask = 1; mask < 16; mask <<= 1)
#pragma unroll
    for (int mi = 0; mi < 2; ++mi)
#pragma unroll
      for (int r = 0; r < 8; ++r)
        rm[mi][r] = fmaxf(rm[mi][r], __shfl_xor(rm[mi][r], mask, 32));
  if ((lane & 15) == 0) {
    const int half8 = (lane >> 4) << 3;
#pragma unroll
    for (int mi = 0; mi < 2; ++mi)
#pragma unroll
      for (int r = 0; r < 8; ++r) redm[m1 + 16 * mi + half8 + r][w >> 1] = rm[mi][r];
  }
  __syncthreads();
  float rs[2][8] = {};
#pragma unroll
  for (int mi = 0; mi < 2; ++mi)
#pragma unroll
    for (int r = 0; r < 8; ++r) {
      const int row = m1 + 16 * mi + ((lane >> 4) << 3) + r;
      rm[mi][r] = fmaxf(fmaxf(redm[row][0], redm[row][1]),
                        fmaxf(redm[row][2], redm[row][3]));
    }
#pragma unroll
  for (int mi = 0; mi < 2; ++mi)
#pragma unroll
    for (int ni = 0; ni < 4; ++ni)
#pragma unroll
      for (int r = 0; r < 8; ++r) {
        const float e = exp2f((acc[mi][ni][r] - rm[mi][r]) * lsc);
        acc[mi][ni][r] = e;
        rs[mi][r] += e;
      }
#pragma unroll
  for (int mask = 1; mask < 16; mask <<= 1)
#pragma unroll
    for (int mi = 0; mi < 2; ++mi)
#pragma unroll
      for (int r = 0; r < 8; ++r) rs[mi][r] += __shfl_xor(rs[mi][r], mask, 32);
  if ((lane & 15) == 0) {
    const int half8 = (lane >> 4) << 3;
#pragma unroll
    for (int mi = 0; mi < 2; ++mi)
#pragma unroll
      for (int r = 0; r < 8; ++r) reds[m1 + 16 * mi + half8 + r][w >> 1] = rs[mi][r];
  }
  __syncthreads();
#pragma unroll
  for (int mi = 0; mi < 2; ++mi)
#pragma unroll
    for (int r = 0; r < 8; ++r) {
      const int row = m1 + 16 * mi + ((lane >> 4) << 3) + r;
      rm[mi][r] = 1.0f / (reds[row][0] + reds[row][1] + reds[row][2] + reds[row][3]);
    }
#pragma unroll
  for (int mi = 0; mi < 2; ++mi)
#pragma unroll
    for (int ni = 0; ni < 4; ++ni) {
      const int col = n1 + 16 * ni + (lane & 15);
#pragma unroll
      for (int r = 0; r < 8; ++r) {
        const int row = m1 + 16 * mi + ((lane >> 4) << 3) + r;
        Ps[row * 264 + col] = (bf16)(acc[mi][ni][r] * rm[mi][r]);
      }
    }
  __syncthreads();

  // ---- GEMM2: O 64x64 = P(64x256) @ V(256x64), waves 2(M) x 4(N16) ----
  const int m2 = (w >> 2) * 32;
  const int n2 = (w & 3) * 16;
  v8f o[2];
  o[0] = zero8(); o[1] = zero8();
#pragma unroll
  for (int ks = 0; ks < 256; ks += 32) {
    v16bf bv = frag_b(Vt, 264, n2, ks);
#pragma unroll
    for (int mi = 0; mi < 2; ++mi) {
      v16bf av = frag_a(Ps, 264, m2 + 16 * mi, ks);
      o[mi] = wmma_bf16(av, bv, o[mi]);
    }
  }
  const size_t oB = (size_t)(b * Sc + s0) * Dc + h * DHc;
  const int col = n2 + (lane & 15);
#pragma unroll
  for (int mi = 0; mi < 2; ++mi)
#pragma unroll
    for (int r = 0; r < 8; ++r) {
      const int row = m2 + 16 * mi + ((lane >> 4) << 3) + r;
      O[oB + (size_t)row * Dc + col] = (bf16)o[mi][r];
    }
}

// ---------------- host orchestration ----------------
extern "C" void kernel_launch(void* const* d_in, const int* in_sizes, int n_in,
                              void* d_out, int out_size, void* d_ws, size_t ws_size,
                              hipStream_t stream) {
  (void)in_sizes; (void)n_in; (void)out_size; (void)ws_size;
  const float* x   = (const float*)d_in[0];
  const float* z   = (const float*)d_in[1];
  const float* wq  = (const float*)d_in[2];  const float* bq  = (const float*)d_in[3];
  const float* wk  = (const float*)d_in[4];  const float* bk  = (const float*)d_in[5];
  const float* wv  = (const float*)d_in[6];  const float* bv  = (const float*)d_in[7];
  const float* Ew  = (const float*)d_in[8];  const float* Eb  = (const float*)d_in[9];
  const float* Fw  = (const float*)d_in[10]; const float* Fb  = (const float*)d_in[11];
  const float* wo  = (const float*)d_in[12]; const float* bo  = (const float*)d_in[13];
  const float* h1w = (const float*)d_in[14]; const float* h1b = (const float*)d_in[15];
  const float* g1w = (const float*)d_in[16]; const float* g1b = (const float*)d_in[17];
  const float* be1w= (const float*)d_in[18]; const float* be1b= (const float*)d_in[19];
  const float* h2w = (const float*)d_in[20]; const float* h2b = (const float*)d_in[21];
  const float* g2w = (const float*)d_in[22]; const float* g2b = (const float*)d_in[23];
  const float* be2w= (const float*)d_in[24]; const float* be2b= (const float*)d_in[25];
  const float* m1w = (const float*)d_in[26]; const float* m1b = (const float*)d_in[27];
  const float* m2w = (const float*)d_in[28]; const float* m2b = (const float*)d_in[29];
  float* out = (float*)d_out;

  char* ws = (char*)d_ws;
  size_t off = 0;
  auto alloc = [&](size_t bytes) -> void* {
    void* p = ws + off;
    off = (off + bytes + 255) & ~(size_t)255;
    return p;
  };
  const size_t MN = (size_t)Bc * Sc;  // 16384 rows

  bf16* wq_b  = (bf16*)alloc((size_t)Dc * Dc * 2);
  bf16* wk_b  = (bf16*)alloc((size_t)Dc * Dc * 2);
  bf16* wv_b  = (bf16*)alloc((size_t)Dc * Dc * 2);
  bf16* wo_b  = (bf16*)alloc((size_t)Dc * Dc * 2);
  bf16* Ew_b  = (bf16*)alloc((size_t)Sc * Kc * 2);
  bf16* Fw_b  = (bf16*)alloc((size_t)Sc * Kc * 2);
  bf16* m1w_b = (bf16*)alloc((size_t)Dc * MLPc * 2);
  bf16* m2w_b = (bf16*)alloc((size_t)MLPc * Dc * 2);
  float* h1   = (float*)alloc((size_t)Bc * Dc * 4);
  float* h2   = (float*)alloc((size_t)Bc * Dc * 4);
  float* sc1  = (float*)alloc((size_t)Bc * Dc * 4);
  float* of1  = (float*)alloc((size_t)Bc * Dc * 4);
  float* sc2  = (float*)alloc((size_t)Bc * Dc * 4);
  float* of2  = (float*)alloc((size_t)Bc * Dc * 4);
  bf16* x1    = (bf16*)alloc(MN * Dc * 2);
  bf16* qb    = (bf16*)alloc(MN * Dc * 2);
  bf16* kb    = (bf16*)alloc(MN * Dc * 2);
  bf16* vb    = (bf16*)alloc(MN * Dc * 2);
  bf16* kp    = (bf16*)alloc((size_t)Bc * Hc * Kc * DHc * 2);
  bf16* vp    = (bf16*)alloc((size_t)Bc * Hc * Kc * DHc * 2);
  bf16* ao    = (bf16*)alloc(MN * Dc * 2);
  float* attn = (float*)alloc(MN * Dc * 4);
  bf16* x2    = (bf16*)alloc(MN * Dc * 2);
  bf16* hmid  = (bf16*)alloc(MN * MLPc * 2);

  auto cvt = [&](const float* src, bf16* dst, size_t n) {
    int n4 = (int)(n / 4);
    cvt_bf16_kernel<<<(n4 + 255) / 256, 256, 0, stream>>>(src, dst, n4);
  };
  cvt(wq, wq_b, (size_t)Dc * Dc);   cvt(wk, wk_b, (size_t)Dc * Dc);
  cvt(wv, wv_b, (size_t)Dc * Dc);   cvt(wo, wo_b, (size_t)Dc * Dc);
  cvt(Ew, Ew_b, (size_t)Sc * Kc);   cvt(Fw, Fw_b, (size_t)Sc * Kc);
  cvt(m1w, m1w_b, (size_t)Dc * MLPc); cvt(m2w, m2w_b, (size_t)MLPc * Dc);

  // conditioning (tiny)
  dim3 sg(Dc / 256, Bc);
  small_dense_kernel<<<sg, 256, 0, stream>>>(z,  h1w, h1b, h1,  ZDc, Dc, 1);
  small_dense_kernel<<<sg, 256, 0, stream>>>(h1, g1w, g1b, sc1, Dc,  Dc, 0);
  small_dense_kernel<<<sg, 256, 0, stream>>>(h1, be1w, be1b, of1, Dc, Dc, 0);
  small_dense_kernel<<<sg, 256, 0, stream>>>(z,  h2w, h2b, h2,  ZDc, Dc, 1);
  small_dense_kernel<<<sg, 256, 0, stream>>>(h2, g2w, g2b, sc2, Dc,  Dc, 0);
  small_dense_kernel<<<sg, 256, 0, stream>>>(h2, be2w, be2b, of2, Dc, Dc, 0);

  // adaLN 1
  adaln_kernel<<<(int)MN, 256, 0, stream>>>(x, sc1, of1, x1);

  // QKV projections (bf16 WMMA)
  dim3 gQ(Dc / 128, (int)MN / 128);
  gemm_bf16_kernel<0><<<gQ, 256, 0, stream>>>(x1, wq_b, bq, nullptr, qb, (int)MN, Dc, Dc);
  gemm_bf16_kernel<0><<<gQ, 256, 0, stream>>>(x1, wk_b, bk, nullptr, kb, (int)MN, Dc, Dc);
  gemm_bf16_kernel<0><<<gQ, 256, 0, stream>>>(x1, wv_b, bv, nullptr, vb, (int)MN, Dc, Dc);

  // Linformer seq projections S->K
  dim3 gE(Kc / 128, Bc * Hc);
  seqproj_kernel<<<gE, 256, 0, stream>>>(Ew_b, kb, Eb, kp);
  seqproj_kernel<<<gE, 256, 0, stream>>>(Fw_b, vb, Fb, vp);

  // fused attention
  dim3 gA(Sc / 64, Bc * Hc);
  attn_kernel<<<gA, 256, 0, stream>>>(qb, kp, vp, ao);

  // output projection + residual (x + attn)
  gemm_bf16_kernel<2><<<gQ, 256, 0, stream>>>(ao, wo_b, bo, x, attn, (int)MN, Dc, Dc);

  // adaLN 2
  adaln_kernel<<<(int)MN, 256, 0, stream>>>(attn, sc2, of2, x2);

  // MLP: gelu(x2@m1w+m1b) @ m2w + m2b + attn
  dim3 g1(MLPc / 128, (int)MN / 128);
  gemm_bf16_kernel<1><<<g1, 256, 0, stream>>>(x2, m1w_b, m1b, nullptr, hmid, (int)MN, MLPc, Dc);
  dim3 g2(Dc / 128, (int)MN / 128);
  gemm_bf16_kernel<2><<<g2, 256, 0, stream>>>(hmid, m2w_b, m2b, attn, out, (int)MN, Dc, MLPc);
}
